// Seq2SeqSubclass_89507118449306
// MI455X (gfx1250) — compile-verified, matrix operand order checked by reading, not run on hardware
//
#include <hip/hip_runtime.h>
#include <hip/hip_bf16.h>

typedef __attribute__((ext_vector_type(16))) _Float16 v16h;
typedef __attribute__((ext_vector_type(8)))  _Float16 v8h;
typedef __attribute__((ext_vector_type(8)))  float    v8f;

#define B_   64
#define TE   1024
#define TD   256
#define FD   8
#define UU   256      // hidden units
#define G4   1024     // 4*UU
#define NH   8
#define DH   32

// ---------------------------------------------------------------------------
// WMMA operand helpers (CDNA5 f16 16x16x32 layouts, wave32)
// A tile (16xK=32, row-major source, ld elements between rows):
//   lane L: row = L%16, holds K halves [kh..kh+7] and [kh+16..kh+23], kh=8*(L>=16)
// B tile (K=32 x 16) uses the mirrored layout: lane L: col = L%16, same K halves,
// so source must be laid out contiguous-in-K (i.e. [N][K]).
// ---------------------------------------------------------------------------
__device__ __forceinline__ v16h wmma_ld_16x32(const _Float16* base, int ld) {
  const int lane = threadIdx.x & 31;
  const int r    = lane & 15;
  const int kh   = (lane >> 4) << 3;      // 0 or 8
  const _Float16* p = base + (long)r * ld + kh;
  v8h lo = *(const v8h*)(p);
  v8h hi = *(const v8h*)(p + 16);
  v16h out;
#pragma unroll
  for (int i = 0; i < 8; ++i) { out[i] = lo[i]; out[i + 8] = hi[i]; }
  return out;
}

// ---------------------------------------------------------------------------
// Prep kernels
// ---------------------------------------------------------------------------
__global__ __launch_bounds__(256) void ln_enc_kernel(
    const float* __restrict__ x, const float* __restrict__ g,
    const float* __restrict__ b, float* __restrict__ y, int rows) {
  int row = blockIdx.x * 256 + threadIdx.x;
  if (row >= rows) return;
  float v[FD], m = 0.f;
#pragma unroll
  for (int j = 0; j < FD; ++j) { v[j] = x[row * FD + j]; m += v[j]; }
  m *= (1.f / FD);
  float var = 0.f;
#pragma unroll
  for (int j = 0; j < FD; ++j) { float d = v[j] - m; var += d * d; }
  var *= (1.f / FD);
  float rstd = rsqrtf(var + 1e-5f);
#pragma unroll
  for (int j = 0; j < FD; ++j) y[row * FD + j] = g[j] * (v[j] - m) * rstd + b[j];
}

// urecT[n][k] (n<1024, k<288): k<256 -> U[k][n]; 256<=k<264 -> W[k-256][n]; else 0
__global__ __launch_bounds__(256) void pack_urec_kernel(
    const float* __restrict__ U, const float* __restrict__ W,
    _Float16* __restrict__ uT) {
  int idx = blockIdx.x * 256 + threadIdx.x;
  if (idx >= G4 * 288) return;
  int n = idx / 288, k = idx % 288;
  float v = 0.f;
  if (k < UU)            v = U[(long)k * G4 + n];
  else if (k < UU + FD)  v = W[(long)(k - UU) * G4 + n];
  uT[idx] = (_Float16)v;
}

// WT[n][k] = Wsrc[k][n]  (256x256)
__global__ __launch_bounds__(256) void pack_w256_kernel(
    const float* __restrict__ Wsrc, _Float16* __restrict__ WT) {
  int idx = blockIdx.x * 256 + threadIdx.x;
  int n = idx >> 8, k = idx & 255;
  WT[idx] = (_Float16)Wsrc[(long)k * 256 + n];
}

// ---------------------------------------------------------------------------
// Persistent dual-LSTM kernel: one 16-wave workgroup, all state in WGP LDS.
//   zS: f16 [1024 cols][64 rows] (128KB)  -- col-major so D-tile scatter packs
//   hS: f16 [64][256]            (32KB)
//   cS: f32 [64*256]             (64KB)   -- cell state, thread-owned
//   xS: f16 [64][32]             (4KB)
//   biasS: f32 [1024]            (4KB)
// ---------------------------------------------------------------------------
#define ZS_OFF   0
#define HS_OFF   131072
#define CS_OFF   (131072 + 32768)
#define XS_OFF   (131072 + 32768 + 65536)
#define BS_OFF   (131072 + 32768 + 65536 + 4096)
#define LSTM_SHMEM (131072 + 32768 + 65536 + 4096 + 4096)

__global__ __launch_bounds__(512) void lstm_seq_kernel(
    const float* __restrict__ lnx,       // [B,TE,8]
    const float* __restrict__ dec_in,    // [B,TD,8]
    const _Float16* __restrict__ urecT_e,// [1024][288]
    const _Float16* __restrict__ urecT_d,
    const float* __restrict__ bias_e,    // [1024]
    const float* __restrict__ bias_d,
    _Float16* __restrict__ enc_seq,      // [B,TE,256]
    _Float16* __restrict__ dec_seq)      // [B,TD,256]
{
  extern __shared__ char smem[];
  _Float16* zS    = (_Float16*)(smem + ZS_OFF);
  _Float16* hS    = (_Float16*)(smem + HS_OFF);
  float*    cS    = (float*)   (smem + CS_OFF);
  _Float16* xS    = (_Float16*)(smem + XS_OFF);
  float*    biasS = (float*)   (smem + BS_OFF);

  const int tid  = threadIdx.x;
  const int wave = tid >> 5;
  const int lane = tid & 31;
  const int col0  = lane & 15;
  const int rbase = (lane >> 4) << 3;

  for (int i = tid; i < B_ * UU; i += 512) { hS[i] = (_Float16)0.f; cS[i] = 0.f; }

  for (int phase = 0; phase < 2; ++phase) {
    const int T = phase ? TD : TE;
    const float*    xin  = phase ? dec_in   : lnx;
    const _Float16* uT   = phase ? urecT_d  : urecT_e;
    const float*    bias = phase ? bias_d   : bias_e;
    _Float16*       seq  = phase ? dec_seq  : enc_seq;

    for (int i = tid; i < G4; i += 512) biasS[i] = bias[i];
    __syncthreads();

    for (int t = 0; t < T; ++t) {
      // stage x_t padded to K=32
      for (int i = tid; i < B_ * 32; i += 512) {
        int bb = i >> 5, j = i & 31;
        float v = (j < FD) ? xin[((long)bb * T + t) * FD + j] : 0.f;
        xS[i] = (_Float16)v;
      }
      __syncthreads();

      // z = [h | x_pad] @ uT^T : each wave owns N columns [wave*64, wave*64+64)
      v8f zero = {};
      v8f acc[4][4];
#pragma unroll
      for (int mt = 0; mt < 4; ++mt)
#pragma unroll
        for (int nt = 0; nt < 4; ++nt) acc[mt][nt] = zero;

      for (int kb = 0; kb < 9; ++kb) {
        v16h a[4];
        if (kb < 8) {
#pragma unroll
          for (int mt = 0; mt < 4; ++mt)
            a[mt] = wmma_ld_16x32(hS + mt * 16 * UU + kb * 32, UU);
        } else {
#pragma unroll
          for (int mt = 0; mt < 4; ++mt)
            a[mt] = wmma_ld_16x32(xS + mt * 16 * 32, 32);
        }
#pragma unroll
        for (int nt = 0; nt < 4; ++nt) {
          const int n0 = (wave * 4 + nt) * 16;
          if (kb < 8)
            __builtin_prefetch((const void*)(uT + (long)n0 * 288 + (kb + 1) * 32), 0, 3);
          v16h bm = wmma_ld_16x32(uT + (long)n0 * 288 + kb * 32, 288);
#pragma unroll
          for (int mt = 0; mt < 4; ++mt)
            acc[mt][nt] = __builtin_amdgcn_wmma_f32_16x16x32_f16(
                false, a[mt], false, bm, (short)0, acc[mt][nt], false, false);
        }
      }

      // scatter D tiles into zS (col-major): 8 rows/lane are contiguous -> b128
#pragma unroll
      for (int nt = 0; nt < 4; ++nt) {
        const int n = (wave * 4 + nt) * 16 + col0;
#pragma unroll
        for (int mt = 0; mt < 4; ++mt) {
          v8h pk;
#pragma unroll
          for (int v = 0; v < 8; ++v) pk[v] = (_Float16)acc[mt][nt][v];
          *(v8h*)(zS + n * B_ + mt * 16 + rbase) = pk;
        }
      }
      __syncthreads();

      // gates (keras order i,f,g,o); cell state lives in LDS, thread-owned
      for (int j = 0; j < 32; ++j) {
        const int p = tid + j * 512;
        const int bb = p >> 8, u = p & 255;
        const float zi = (float)zS[(u)         * B_ + bb] + biasS[u];
        const float zf = (float)zS[(UU + u)    * B_ + bb] + biasS[UU + u];
        const float zg = (float)zS[(2*UU + u)  * B_ + bb] + biasS[2*UU + u];
        const float zo = (float)zS[(3*UU + u)  * B_ + bb] + biasS[3*UU + u];
        const float ig = 1.f / (1.f + __expf(-zi));
        const float fg = 1.f / (1.f + __expf(-zf));
        const float og = 1.f / (1.f + __expf(-zo));
        const float cc = fg * cS[p] + ig * tanhf(zg);
        cS[p] = cc;
        const float h = og * tanhf(cc);
        hS[p] = (_Float16)h;
        seq[((long)bb * T + t) * UU + u] = (_Float16)h;
      }
      __syncthreads();
    }
  }
}

// ---------------------------------------------------------------------------
// Generic 256-K GEMM: C[M,256] = A[M,256](f16) @ WT[n][k](f16) + bias
// mode 0: out f16 row-major; mode 1: out f16 head-transposed [b][n][Tk];
// mode 2: out f32 row-major.  One 16x16 tile per wave. Tk = 1<<tksh.
// ---------------------------------------------------------------------------
__global__ __launch_bounds__(256) void gemm256_kernel(
    const _Float16* __restrict__ A, const _Float16* __restrict__ WT,
    const float* __restrict__ bias, int M, int mode, int tksh,
    _Float16* __restrict__ outh, float* __restrict__ outf)
{
  const int wave = threadIdx.x >> 5, lane = threadIdx.x & 31;
  const long tile = (long)blockIdx.x * 8 + wave;
  const int mt = (int)(tile >> 4);
  const int nt = (int)(tile & 15);
  if (mt * 16 >= M) return;

  v8f acc = {};
#pragma unroll
  for (int kb = 0; kb < 8; ++kb) {
    v16h a = wmma_ld_16x32(A  + (long)mt * 16 * 256 + kb * 32, 256);
    v16h b = wmma_ld_16x32(WT + (long)nt * 16 * 256 + kb * 32, 256);
    acc = __builtin_amdgcn_wmma_f32_16x16x32_f16(
        false, a, false, b, (short)0, acc, false, false);
  }
  const int col   = nt * 16 + (lane & 15);
  const int rbase = mt * 16 + ((lane >> 4) << 3);
  const float bv  = bias ? bias[col] : 0.f;
#pragma unroll
  for (int v = 0; v < 8; ++v) {
    const long m = rbase + v;
    const float val = acc[v] + bv;
    if (mode == 0) {
      outh[m * 256 + col] = (_Float16)val;
    } else if (mode == 1) {
      const long bb = m >> tksh;
      const long t  = m & ((1L << tksh) - 1);
      outh[((bb * 256 + col) << tksh) + t] = (_Float16)val;   // [b][n][t]
    } else {
      outf[m * 256 + col] = val;
    }
  }
}

// ---------------------------------------------------------------------------
// Attention: one block per (b, head, 16-row q-tile). 128 threads (4 waves).
// S[16][1024] f32 (64KB) + P[16][1024] f16 (32KB) + inv[16] in dynamic LDS.
// ---------------------------------------------------------------------------
#define ATTN_SHMEM (65536 + 32768 + 64)

__global__ __launch_bounds__(128) void attn_kernel(
    const _Float16* __restrict__ Q,   // [B][Tq][256]
    const _Float16* __restrict__ K,   // [B][Tk][256]
    const _Float16* __restrict__ VT,  // [B][256][Tk]
    _Float16* __restrict__ ctx)       // [B][Tq][256]
{
  extern __shared__ char smem[];
  float*    S    = (float*)smem;                       // 16 x 1024
  _Float16* P    = (_Float16*)(smem + 65536);          // 16 x 1024
  float*    invS = (float*)(smem + 65536 + 32768);     // 16

  const int bid = blockIdx.x;
  const int qt = bid & 15, hh = (bid >> 4) & 7, bb = bid >> 7;
  const int wave = threadIdx.x >> 5, lane = threadIdx.x & 31;

  const _Float16* Qp  = Q  + ((long)bb * TD + qt * 16) * 256 + hh * DH;
  const _Float16* Kp  = K  + (long)bb * TE * 256 + hh * DH;
  const _Float16* VTp = VT + ((long)bb * 256 + hh * DH) * TE;

  // scores: single K=32 WMMA per 16x16 tile
  const v16h aq = wmma_ld_16x32(Qp, 256);
  const int col0 = lane & 15, rbase = (lane >> 4) << 3;
  for (int kt = wave; kt < TE / 16; kt += 4) {
    v16h kb = wmma_ld_16x32(Kp + (long)kt * 16 * 256, 256);
    v8f s = {};
    s = __builtin_amdgcn_wmma_f32_16x16x32_f16(
        false, aq, false, kb, (short)0, s, false, false);
    const int col = kt * 16 + col0;
#pragma unroll
    for (int v = 0; v < 8; ++v)
      S[(rbase + v) * TE + col] = s[v] * 0.17677669529663687f;  // 1/sqrt(32)
  }
  __syncthreads();

  // softmax over 1024: 8 threads per q-row
  const int row = threadIdx.x >> 3, sub = threadIdx.x & 7;
  float mx = -1e30f;
  for (int j = sub; j < TE; j += 8) mx = fmaxf(mx, S[row * TE + j]);
#pragma unroll
  for (int m = 4; m >= 1; m >>= 1) mx = fmaxf(mx, __shfl_xor(mx, m, 32));
  float sum = 0.f;
  for (int j = sub; j < TE; j += 8) {
    float e = __expf(S[row * TE + j] - mx);
    sum += e;
    P[row * TE + j] = (_Float16)e;
  }
#pragma unroll
  for (int m = 4; m >= 1; m >>= 1) sum += __shfl_xor(sum, m, 32);
  if (sub == 0) invS[row] = 1.f / sum;
  __syncthreads();

  // ctx = P[16,1024] @ V[1024,32]; waves split K-blocks, reduce via LDS
  v8f cacc[2] = {{}, {}};
  for (int kb = wave; kb < TE / 32; kb += 4) {
    v16h ap = wmma_ld_16x32(P + kb * 32, TE);
#pragma unroll
    for (int nt = 0; nt < 2; ++nt) {
      v16h bv = wmma_ld_16x32(VTp + (long)nt * 16 * TE + kb * 32, TE);
      cacc[nt] = __builtin_amdgcn_wmma_f32_16x16x32_f16(
          false, ap, false, bv, (short)0, cacc[nt], false, false);
    }
  }
  float* part = S;  // reuse score buffer: part[(wave*2+nt)*16 + r][cidx]
#pragma unroll
  for (int nt = 0; nt < 2; ++nt)
#pragma unroll
    for (int v = 0; v < 8; ++v)
      part[((wave * 2 + nt) * 16 + rbase + v) * 16 + col0] = cacc[nt][v];
  __syncthreads();

  for (int idx = threadIdx.x; idx < 2 * 16 * 16; idx += 128) {
    const int nt = idx >> 8, r = (idx >> 4) & 15, cc = idx & 15;
    float acc = 0.f;
#pragma unroll
    for (int w = 0; w < 4; ++w) acc += part[((w * 2 + nt) * 16 + r) * 16 + cc];
    acc *= invS[r];
    ctx[((long)bb * TD + qt * 16 + r) * 256 + hh * DH + nt * 16 + cc] = (_Float16)acc;
  }
}

// ---------------------------------------------------------------------------
// Residual + LayerNorm + out projection. One wave per (b,t) row.
// ---------------------------------------------------------------------------
__global__ __launch_bounds__(256) void mix_out_kernel(
    const _Float16* __restrict__ dec_seq, const float* __restrict__ attnout,
    const float* __restrict__ g, const float* __restrict__ bta,
    const float* __restrict__ W_out, const float* __restrict__ b_out,
    float* __restrict__ out)
{
  const int wave = threadIdx.x >> 5, lane = threadIdx.x & 31;
  const long row = (long)blockIdx.x * 8 + wave;
  float vals[8], mean = 0.f;
#pragma unroll
  for (int j = 0; j < 8; ++j) {
    const int col = lane + j * 32;
    const float x = (float)dec_seq[row * 256 + col] + attnout[row * 256 + col];
    vals[j] = x; mean += x;
  }
#pragma unroll
  for (int m = 16; m >= 1; m >>= 1) mean += __shfl_xor(mean, m, 32);
  mean *= (1.f / 256.f);
  float var = 0.f;
#pragma unroll
  for (int j = 0; j < 8; ++j) { float d = vals[j] - mean; var += d * d; }
#pragma unroll
  for (int m = 16; m >= 1; m >>= 1) var += __shfl_xor(var, m, 32);
  const float rstd = rsqrtf(var * (1.f / 256.f) + 1e-5f);
  float acc = 0.f;
#pragma unroll
  for (int j = 0; j < 8; ++j) {
    const int col = lane + j * 32;
    const float y = g[col] * (vals[j] - mean) * rstd + bta[col];
    acc += y * W_out[col];
  }
#pragma unroll
  for (int m = 16; m >= 1; m >>= 1) acc += __shfl_xor(acc, m, 32);
  if (lane == 0) out[row] = acc + b_out[0];
}

// ---------------------------------------------------------------------------
extern "C" void kernel_launch(void* const* d_in, const int* in_sizes, int n_in,
                              void* d_out, int out_size, void* d_ws, size_t ws_size,
                              hipStream_t stream) {
  const float* enc_in   = (const float*)d_in[0];
  const float* dec_in   = (const float*)d_in[1];
  const float* ln_enc_g = (const float*)d_in[2];
  const float* ln_enc_b = (const float*)d_in[3];
  const float* W_enc    = (const float*)d_in[4];
  const float* U_enc    = (const float*)d_in[5];
  const float* b_enc    = (const float*)d_in[6];
  const float* W_dec    = (const float*)d_in[7];
  const float* U_dec    = (const float*)d_in[8];
  const float* b_dec    = (const float*)d_in[9];
  const float* Wq = (const float*)d_in[10]; const float* bq = (const float*)d_in[11];
  const float* Wk = (const float*)d_in[12]; const float* bk = (const float*)d_in[13];
  const float* Wv = (const float*)d_in[14]; const float* bv = (const float*)d_in[15];
  const float* Wo = (const float*)d_in[16]; const float* bo = (const float*)d_in[17];
  const float* ln_post_g = (const float*)d_in[18];
  const float* ln_post_b = (const float*)d_in[19];
  const float* W_out = (const float*)d_in[20];
  const float* b_out = (const float*)d_in[21];
  float* out = (float*)d_out;
  (void)in_sizes; (void)n_in; (void)out_size; (void)ws_size;

  char* ws = (char*)d_ws;
  size_t o = 0;
  auto carve = [&](size_t bytes) { void* p = ws + o; o = (o + bytes + 255) & ~(size_t)255; return p; };
  float*    lnx      = (float*)   carve((size_t)B_ * TE * FD * 4);
  _Float16* urecT_e  = (_Float16*)carve((size_t)G4 * 288 * 2);
  _Float16* urecT_d  = (_Float16*)carve((size_t)G4 * 288 * 2);
  _Float16* WqT      = (_Float16*)carve(256 * 256 * 2);
  _Float16* WkT      = (_Float16*)carve(256 * 256 * 2);
  _Float16* WvT      = (_Float16*)carve(256 * 256 * 2);
  _Float16* WoT      = (_Float16*)carve(256 * 256 * 2);
  _Float16* enc_seq  = (_Float16*)carve((size_t)B_ * TE * UU * 2);
  _Float16* dec_seq  = (_Float16*)carve((size_t)B_ * TD * UU * 2);
  _Float16* Qbuf     = (_Float16*)carve((size_t)B_ * TD * 256 * 2);
  _Float16* Kbuf     = (_Float16*)carve((size_t)B_ * TE * 256 * 2);
  _Float16* VTbuf    = (_Float16*)carve((size_t)B_ * 256 * TE * 2);
  _Float16* ctxbuf   = (_Float16*)carve((size_t)B_ * TD * 256 * 2);
  float*    attnout  = (float*)   carve((size_t)B_ * TD * 256 * 4);

  (void)hipFuncSetAttribute((const void*)lstm_seq_kernel,
      hipFuncAttributeMaxDynamicSharedMemorySize, LSTM_SHMEM);
  (void)hipFuncSetAttribute((const void*)attn_kernel,
      hipFuncAttributeMaxDynamicSharedMemorySize, ATTN_SHMEM);

  // prep
  ln_enc_kernel<<<(B_ * TE + 255) / 256, 256, 0, stream>>>(
      enc_in, ln_enc_g, ln_enc_b, lnx, B_ * TE);
  pack_urec_kernel<<<(G4 * 288 + 255) / 256, 256, 0, stream>>>(U_enc, W_enc, urecT_e);
  pack_urec_kernel<<<(G4 * 288 + 255) / 256, 256, 0, stream>>>(U_dec, W_dec, urecT_d);
  pack_w256_kernel<<<256, 256, 0, stream>>>(Wq, WqT);
  pack_w256_kernel<<<256, 256, 0, stream>>>(Wk, WkT);
  pack_w256_kernel<<<256, 256, 0, stream>>>(Wv, WvT);
  pack_w256_kernel<<<256, 256, 0, stream>>>(Wo, WoT);

  // persistent dual LSTM (encoder then decoder) on one WGP, state in LDS
  lstm_seq_kernel<<<1, 512, LSTM_SHMEM, stream>>>(
      lnx, dec_in, urecT_e, urecT_d, b_enc, b_dec, enc_seq, dec_seq);

  // projections
  gemm256_kernel<<<(B_ * TD / 16) * 16 / 8, 256, 0, stream>>>(
      dec_seq, WqT, bq, B_ * TD, 0, 8, Qbuf, nullptr);
  gemm256_kernel<<<(B_ * TE / 16) * 16 / 8, 256, 0, stream>>>(
      enc_seq, WkT, bk, B_ * TE, 0, 10, Kbuf, nullptr);
  gemm256_kernel<<<(B_ * TE / 16) * 16 / 8, 256, 0, stream>>>(
      enc_seq, WvT, bv, B_ * TE, 1, 10, VTbuf, nullptr);

  // attention
  attn_kernel<<<B_ * NH * (TD / 16), 128, ATTN_SHMEM, stream>>>(
      Qbuf, Kbuf, VTbuf, ctxbuf);

  // output projection of context
  gemm256_kernel<<<(B_ * TD / 16) * 16 / 8, 256, 0, stream>>>(
      ctxbuf, WoT, bo, B_ * TD, 2, 8, nullptr, attnout);

  // residual + LN + final 256->1 projection
  mix_out_kernel<<<B_ * TD / 8, 256, 0, stream>>>(
      dec_seq, attnout, ln_post_g, ln_post_b, W_out, b_out, out);
}
